// LSTMModel_69733089017933
// MI455X (gfx1250) — matrix operand order repaired and checked by reading, not compiled, not run on hardware
//
#include <hip/hip_runtime.h>

// ------------------------------------------------------------------
// Persistent fused 2-layer LSTM + FC for MI455X (gfx1250, wave32).
// grid = 32 workgroups (16 batch rows each), block = 256 (8 waves).
// Wave w owns hidden units [16w,16w+16): its 4 gate tiles (i,f,g,o).
// All GEMMs via v_wmma_f32_16x16x32_bf16 (f32 accumulate).
// Weight residency: W_hh0 + W_ih0 in VGPRs (layer-0 chain head),
// W_ih1 + W_hh1 as B-tile images in LDS (~256KB, spec allows 320KB).
// h/c state never leaves the WGP: h images live in LDS, c in VGPRs.
// ------------------------------------------------------------------

typedef __attribute__((ext_vector_type(8)))  float        v8f;
typedef __attribute__((ext_vector_type(8)))  unsigned int v8u;
typedef __attribute__((ext_vector_type(16))) __bf16       v16bf;

#define T_STEPS 1024
#define BATCH   512

// ---- LDS layout (bytes). A/B tile image = 32 lanes x 32B = 1KB ----
#define LDS_WIH1 0                         // 32 ntiles * 4 kblocks = 128KB
#define LDS_WHH1 (128 * 1024)              // 128KB
#define LDS_XA   (256 * 1024)              // 1KB  : x_t A-image (16x32, K>=16 zero)
#define LDS_H0A  (LDS_XA + 1024)           // 4KB  : h0 A-images (16x128, 4 kblocks)
#define LDS_H1A  (LDS_H0A + 4096)          // 4KB  : h1 A-images
#define LDS_OUT  (LDS_H1A + 4096)          // 64B  : FC accumulator [16]
#define LDS_SZ   (LDS_OUT + 64)            // ~265KB total (< 320KB WGP limit)

__device__ __forceinline__ unsigned short f2bf(float f) {
  unsigned u = __builtin_bit_cast(unsigned, f);
  u += 0x7fffu + ((u >> 16) & 1u);         // round-to-nearest-even
  return (unsigned short)(u >> 16);
}
__device__ __forceinline__ unsigned pk2(float a, float b) {
  return (unsigned)f2bf(a) | ((unsigned)f2bf(b) << 16);
}
__device__ __forceinline__ float sigf(float x) {
  return 1.0f / (1.0f + __expf(-x));       // v_exp_f32 path
}
__device__ __forceinline__ float tanhfast(float x) {
  return 1.0f - 2.0f / (__expf(2.0f * x) + 1.0f);
}

__device__ __forceinline__ v8u lds_tile(const char* smem, int byteoff, int lane) {
  const uint4* p = (const uint4*)(smem + byteoff + lane * 32);
  uint4 lo = p[0];
  uint4 hi = p[1];
  v8u r;
  r[0] = lo.x; r[1] = lo.y; r[2] = lo.z; r[3] = lo.w;
  r[4] = hi.x; r[5] = hi.y; r[6] = hi.z; r[7] = hi.w;
  return r;
}

__device__ __forceinline__ v8f wmma_bf16(v8u a, v8u b, v8f c) {
  return __builtin_amdgcn_wmma_f32_16x16x32_bf16(
      false, __builtin_bit_cast(v16bf, a),
      false, __builtin_bit_cast(v16bf, b),
      (short)0, c, false, false);
}

__global__ __launch_bounds__(256, 1)
void lstm2_fused(const float* __restrict__ x,
                 const float* __restrict__ Wih0, const float* __restrict__ Whh0,
                 const float* __restrict__ bi0,  const float* __restrict__ bh0,
                 const float* __restrict__ Wih1, const float* __restrict__ Whh1,
                 const float* __restrict__ bi1,  const float* __restrict__ bh1,
                 const float* __restrict__ Wfc,  const float* __restrict__ bfc,
                 float* __restrict__ out)
{
  __shared__ __align__(16) char smem[LDS_SZ];
  const int tid  = threadIdx.x;
  const int wave = tid >> 5;
  const int lane = tid & 31;
  const int ln   = lane & 15;
  const int lh   = lane >> 4;              // 0: lanes 0-15, 1: lanes 16-31
  const int bb   = blockIdx.x * 16;        // batch tile base

  // ------- per-wave register weight tiles (WMMA B layout, bf16) -------
  // B 32x16, contiguous K order used here: lane<16 -> N=lane, K=0..15 seq; lane>=16 -> N=lane-16, K=16..31
  v8u whh0[4][4], wih0t[4];
  float bias0[4], bias1[4];
#pragma unroll
  for (int g = 0; g < 4; ++g) {
    const int n = (g * 8 + wave) * 16 + ln;     // gate column this lane holds
    bias0[g] = bi0[n] + bh0[n];
    bias1[g] = bi1[n] + bh1[n];
    {   // W_ih0: K real = 0..15, upper half of K padded with zeros
      v8u t;
      if (lh == 0) {
        const float* p = Wih0 + n * 16;
#pragma unroll
        for (int j = 0; j < 8; ++j) t[j] = pk2(p[2 * j], p[2 * j + 1]);
      } else {
#pragma unroll
        for (int j = 0; j < 8; ++j) t[j] = 0u;
      }
      wih0t[g] = t;
    }
#pragma unroll
    for (int kb = 0; kb < 4; ++kb) {
      const float* p0 = Whh0 + n * 128 + kb * 32 + lh * 16;
      v8u t0;
#pragma unroll
      for (int j = 0; j < 8; ++j) t0[j] = pk2(p0[2 * j], p0[2 * j + 1]);
      whh0[g][kb] = t0;
    }
  }
  const float wfc = Wfc[wave * 16 + ln];
  const float bfb = bfc[0];

  // ------- stage W_ih1 (imgs 0..127) and W_hh1 (imgs 128..255) B-tiles -------
  for (int img = wave; img < 256; img += 8) {
    const int m2 = img & 127;
    const int nt = m2 >> 2, kb = m2 & 3;
    const int n  = nt * 16 + ln;
    const float* p = (img < 128 ? Wih1 : Whh1) + n * 128 + kb * 32 + lh * 16;
    unsigned* dst = (unsigned*)(smem + img * 1024 + lane * 32);
#pragma unroll
    for (int j = 0; j < 8; ++j) dst[j] = pk2(p[2 * j], p[2 * j + 1]);
  }
  // zero x/h A-images (initial h=0; xA's K>=16 half stays 0 forever)
  {
    unsigned* z = (unsigned*)(smem + LDS_XA);
    for (int i = tid; i < (1024 + 4096 + 4096) / 4; i += 256) z[i] = 0u;
  }
  __syncthreads();

  v8f c0, c1;
#pragma unroll
  for (int e = 0; e < 8; ++e) { c0[e] = 0.f; c1[e] = 0.f; }

  float* outstg = (float*)(smem + LDS_OUT);

  // scatter coordinates for writing this wave's h slice back into the
  // shared A-images (contiguous-K A layout used here, matching the B tiles: K>=16 in hi lanes;
  // byte = (K&15)*2)
  const int j_hid = wave * 16 + ln;
  const int kb_h  = j_hid >> 5;
  const int K_h   = j_hid & 31;
  const int g2_h  = K_h >> 3;
  const int dlane = (K_h >> 4) * 16;
  const int boff  = (K_h & 15) * 2;

#pragma clang loop unroll(disable)
  for (int t = 0; t < T_STEPS; ++t) {
    // stage x_t A-image (wave 0); zero FC accumulator
    if (wave == 0) {
      const float* xp = x + ((size_t)(bb + ln) * T_STEPS + t) * 16;
      unsigned* dst = (unsigned*)(smem + LDS_XA + lane * 32);
#pragma unroll
      for (int j = 0; j < 8; ++j) if (lh == 0) dst[j] = pk2(xp[2 * j], xp[2 * j + 1]);
    }
    if (tid < 16) outstg[tid] = 0.f;
    __syncthreads();

    // ---- layer 0: gates = x_t*Wih0^T + h0_{t-1}*Whh0^T + b ----
    v8f acc[4];
#pragma unroll
    for (int g = 0; g < 4; ++g)
#pragma unroll
      for (int e = 0; e < 8; ++e) acc[g][e] = bias0[g];
    {
      v8u a = lds_tile(smem, LDS_XA, lane);
#pragma unroll
      for (int g = 0; g < 4; ++g) acc[g] = wmma_bf16(a, wih0t[g], acc[g]);
    }
#pragma unroll
    for (int kb = 0; kb < 4; ++kb) {
      v8u a = lds_tile(smem, LDS_H0A + kb * 1024, lane);
#pragma unroll
      for (int g = 0; g < 4; ++g) acc[g] = wmma_bf16(a, whh0[g][kb], acc[g]);
    }
    v8f h0;
#pragma unroll
    for (int e = 0; e < 8; ++e) {
      float ig = sigf(acc[0][e]);
      float fg = sigf(acc[1][e]);
      float gg = tanhfast(acc[2][e]);
      float og = sigf(acc[3][e]);
      float c  = fg * c0[e] + ig * gg;
      c0[e] = c;
      h0[e] = og * tanhfast(c);
    }
    __syncthreads();                       // all waves done reading h0A(t-1)

    {   // scatter h0_t -> h0 A-images (8 x ds_store_b16 per lane)
      char* bp = smem + LDS_H0A + kb_h * 1024 + boff;
#pragma unroll
      for (int e = 0; e < 8; ++e) {
        const int m = e + lh * 8;
        *(unsigned short*)(bp + (m + dlane) * 32) = f2bf(h0[e]);
      }
    }
    __syncthreads();                       // h0A(t) visible to all waves

    // ---- layer 1: gates = h0_t*Wih1^T + h1_{t-1}*Whh1^T + b ----
    v8f acc2[4];
#pragma unroll
    for (int g = 0; g < 4; ++g)
#pragma unroll
      for (int e = 0; e < 8; ++e) acc2[g][e] = bias1[g];
#pragma unroll
    for (int kb = 0; kb < 4; ++kb) {
      v8u a = lds_tile(smem, LDS_H0A + kb * 1024, lane);
#pragma unroll
      for (int g = 0; g < 4; ++g) {
        v8u b = lds_tile(smem, LDS_WIH1 + ((g * 8 + wave) * 4 + kb) * 1024, lane);
        acc2[g] = wmma_bf16(a, b, acc2[g]);
      }
    }
#pragma unroll
    for (int kb = 0; kb < 4; ++kb) {
      v8u a = lds_tile(smem, LDS_H1A + kb * 1024, lane);
#pragma unroll
      for (int g = 0; g < 4; ++g) {
        v8u b = lds_tile(smem, LDS_WHH1 + ((g * 8 + wave) * 4 + kb) * 1024, lane);
        acc2[g] = wmma_bf16(a, b, acc2[g]);
      }
    }
    v8f h1;
#pragma unroll
    for (int e = 0; e < 8; ++e) {
      float ig = sigf(acc2[0][e]);
      float fg = sigf(acc2[1][e]);
      float gg = tanhfast(acc2[2][e]);
      float og = sigf(acc2[3][e]);
      float c  = fg * c1[e] + ig * gg;
      c1[e] = c;
      h1[e] = og * tanhfast(c);
    }
    __syncthreads();                       // all waves done reading h1A(t-1)

    {   // scatter h1_t -> h1 A-images; fused FC: out += relu(h1)*wfc
      char* bp = smem + LDS_H1A + kb_h * 1024 + boff;
#pragma unroll
      for (int e = 0; e < 8; ++e) {
        const int m = e + lh * 8;
        *(unsigned short*)(bp + (m + dlane) * 32) = f2bf(h1[e]);
        float r = h1[e] > 0.f ? h1[e] : 0.f;
        atomicAdd(&outstg[m], r * wfc);    // ds_add_f32
      }
    }
    __syncthreads();

    if (tid < 16) {
      float v = outstg[tid] + bfb;
      out[(size_t)(bb + tid) * T_STEPS + t] = v > 0.f ? v : 0.f;
    }
  }
}

extern "C" void kernel_launch(void* const* d_in, const int* in_sizes, int n_in,
                              void* d_out, int out_size, void* d_ws, size_t ws_size,
                              hipStream_t stream) {
  (void)in_sizes; (void)n_in; (void)d_ws; (void)ws_size; (void)out_size;
  const float* xp   = (const float*)d_in[0];
  const float* Wih0 = (const float*)d_in[1];
  const float* Whh0 = (const float*)d_in[2];
  const float* bi0  = (const float*)d_in[3];
  const float* bh0  = (const float*)d_in[4];
  const float* Wih1 = (const float*)d_in[5];
  const float* Whh1 = (const float*)d_in[6];
  const float* bi1  = (const float*)d_in[7];
  const float* bh1  = (const float*)d_in[8];
  const float* Wfc  = (const float*)d_in[9];
  const float* bfc  = (const float*)d_in[10];
  lstm2_fused<<<dim3(BATCH / 16), dim3(256), 0, stream>>>(
      xp, Wih0, Whh0, bi0, bh0, Wih1, Whh1, bi1, bh1, Wfc, bfc, (float*)d_out);
}
